// MultiheadAttention_10986526343534
// MI455X (gfx1250) — matrix operand order repaired and checked by reading, not compiled, not run on hardware
//
#include <hip/hip_runtime.h>

// ---------------------------------------------------------------------------
// MHA forward for MI455X (gfx1250): bf16 WMMA (16x16x32), f32 accumulate.
// convert -> QKV GEMM (V stored transposed) -> flash attention -> out GEMM.
// GEMM/attention loops use ping-pong (unroll-2) double buffering, no rotate
// copies; __launch_bounds__(128,1) unlocks full VGPR budget (no spills).
// ---------------------------------------------------------------------------

typedef __bf16 v16bf __attribute__((ext_vector_type(16)));
typedef __bf16 v8bf  __attribute__((ext_vector_type(8)));
typedef float  v8f   __attribute__((ext_vector_type(8)));

#define WMMA_BF16(a, b, c) \
    __builtin_amdgcn_wmma_f32_16x16x32_bf16(false, (a), false, (b), (short)0, (c), false, false)

constexpr int Bz = 2;         // batch
constexpr int S  = 2048;      // sequence
constexpr int D  = 1024;      // model dim
constexpr int H  = 16;        // heads
constexpr int HD = 64;        // head dim
constexpr int BS = Bz * S;    // 4096 rows
constexpr int D3 = 3 * D;     // 3072 qkv channels

// ---------------------------------------------------------------------------
// Fragment loaders (wave32 WMMA fragments; lane half h holds k = 16h..16h+15 in both A and B)
// ---------------------------------------------------------------------------

// A-matrix 16x32 bf16, source row-major [rows x ld], tile at (row0, col0).
static __device__ __forceinline__ v16bf load_a_frag(const __bf16* base, int ld,
                                                    int row0, int col0, int lane) {
    const int off = (lane >> 4) << 4;  // 0 or 16
    const __bf16* p = base + (size_t)(row0 + (lane & 15)) * ld + col0 + off;
    v16bf a;
#pragma unroll
    for (int e = 0; e < 16; ++e) {
        a[e]     = p[e];
    }
    return a;
}

// B-matrix 32x16 bf16 where B[k][n] = W[n][k], W row-major [N x ld].
// Per-lane read: 16 contiguous bf16 (32 bytes).
static __device__ __forceinline__ v16bf load_bT_frag(const __bf16* W, int ld,
                                                     int n0, int k0, int lane) {
    const int off = (lane >> 4) << 4;  // 0 or 16
    const __bf16* p = W + (size_t)(n0 + (lane & 15)) * ld + k0 + off;
    v16bf b;
#pragma unroll
    for (int e = 0; e < 16; ++e) b[e] = p[e];
    return b;
}

// ---------------------------------------------------------------------------
// Kernel 1: f32 -> bf16 convert, 8 elements / thread (b128 traffic)
// ---------------------------------------------------------------------------
__global__ void cvt_f32_bf16(const float* __restrict__ src,
                             __bf16* __restrict__ dst, int n8) {
    const int i = blockIdx.x * blockDim.x + threadIdx.x;
    if (i < n8) {
        v8f x = *(const v8f*)(src + (size_t)i * 8);
        v8bf y;
#pragma unroll
        for (int e = 0; e < 8; ++e) y[e] = (__bf16)x[e];
        *(v8bf*)(dst + (size_t)i * 8) = y;
    }
}

// ---------------------------------------------------------------------------
// 32x64 GEMM mainloop with ping-pong double buffering (two fragment sets,
// two 32-deep K chunks per iteration, no rotate copies).
// Kdim must be a multiple of 64.
// ---------------------------------------------------------------------------
struct FragSet { v16bf a0, a1, b0, b1, b2, b3; };

static __device__ __forceinline__ void load_set(FragSet& f,
                                                const __bf16* __restrict__ A, int lda, int m0,
                                                const __bf16* __restrict__ W, int ldw, int n0,
                                                int kk, int lane) {
    f.a0 = load_a_frag(A, lda, m0,      kk, lane);
    f.a1 = load_a_frag(A, lda, m0 + 16, kk, lane);
    f.b0 = load_bT_frag(W, ldw, n0,      kk, lane);
    f.b1 = load_bT_frag(W, ldw, n0 + 16, kk, lane);
    f.b2 = load_bT_frag(W, ldw, n0 + 32, kk, lane);
    f.b3 = load_bT_frag(W, ldw, n0 + 48, kk, lane);
}

static __device__ __forceinline__ void mma_set(const FragSet& f, v8f acc[2][4]) {
    acc[0][0] = WMMA_BF16(f.a0, f.b0, acc[0][0]);
    acc[1][0] = WMMA_BF16(f.a1, f.b0, acc[1][0]);
    acc[0][1] = WMMA_BF16(f.a0, f.b1, acc[0][1]);
    acc[1][1] = WMMA_BF16(f.a1, f.b1, acc[1][1]);
    acc[0][2] = WMMA_BF16(f.a0, f.b2, acc[0][2]);
    acc[1][2] = WMMA_BF16(f.a1, f.b2, acc[1][2]);
    acc[0][3] = WMMA_BF16(f.a0, f.b3, acc[0][3]);
    acc[1][3] = WMMA_BF16(f.a1, f.b3, acc[1][3]);
}

static __device__ __forceinline__ void gemm_mainloop(const __bf16* __restrict__ A, int lda, int m0,
                                                     const __bf16* __restrict__ W, int ldw, int n0,
                                                     int Kdim, int lane, v8f acc[2][4]) {
    FragSet s0, s1;
    load_set(s0, A, lda, m0, W, ldw, n0, 0, lane);
    for (int kk = 0; kk < Kdim - 64; kk += 64) {
        load_set(s1, A, lda, m0, W, ldw, n0, kk + 32, lane);
        mma_set(s0, acc);
        load_set(s0, A, lda, m0, W, ldw, n0, kk + 64, lane);
        mma_set(s1, acc);
    }
    load_set(s1, A, lda, m0, W, ldw, n0, Kdim - 32, lane);
    mma_set(s0, acc);
    mma_set(s1, acc);
}

// ---------------------------------------------------------------------------
// Kernel 2: QKV projection.  One wave computes a 32x64 tile of x @ W_qkv^T,
// adds bias, scatters bf16 into Q,K buffers [B,H,S,HD] and V TRANSPOSED
// buffer [B,H,HD,S] (so attention's P@V B-fragments are contiguous reads).
// ---------------------------------------------------------------------------
__global__ void __launch_bounds__(128, 1)
qkv_gemm(const __bf16* __restrict__ xb,
         const __bf16* __restrict__ wqkv,
         const float* __restrict__ bias,
         __bf16* __restrict__ qb,
         __bf16* __restrict__ kb,
         __bf16* __restrict__ vtb) {
    const int lane = threadIdx.x & 31;
    const int wave = (blockIdx.x * blockDim.x + threadIdx.x) >> 5;
    const int NT = D3 / 64;                 // 48
    const int mt = wave / NT;               // 0..127
    const int nt = wave % NT;               // 0..47
    const int m0 = mt * 32, n0 = nt * 64;

    v8f acc[2][4] = {};
    gemm_mainloop(xb, D, m0, wqkv, D, n0, D, lane, acc);

    const int half = lane >> 4, ln = lane & 15;
#pragma unroll
    for (int t = 0; t < 4; ++t) {
        const int e = n0 + 16 * t + ln;       // output channel [0, 3072)
        const float bv = bias[e];
        const int h = e / 192, j = e % 192;
        const int which = j / 64, hd = j % 64;   // which uniform across lanes
#pragma unroll
        for (int msub = 0; msub < 2; ++msub) {
#pragma unroll
            for (int r = 0; r < 8; ++r) {
                const int row = m0 + msub * 16 + r + half * 8;   // [0, BS)
                const int b = row / S, s = row % S;
                const __bf16 val = (__bf16)(acc[msub][t][r] + bv);
                if (which == 0)
                    qb[(((size_t)(b * H + h)) * S + s) * HD + hd] = val;
                else if (which == 1)
                    kb[(((size_t)(b * H + h)) * S + s) * HD + hd] = val;
                else
                    vtb[(((size_t)(b * H + h)) * HD + hd) * S + s] = val;
            }
        }
    }
}

// ---------------------------------------------------------------------------
// Kernel 3: flash attention.  One wave per (b, h, 16-row q-tile).
// Processes one 32-key chunk; V^T fragments are loaded at chunk start and
// consumed only after softmax (intra-chunk overlap).  K fragments for the
// other ping-pong chunk are prefetched by the caller loop.
// ---------------------------------------------------------------------------
static __device__ __forceinline__ void attn_chunk(const __bf16* __restrict__ Vt, int kk,
                                                  int lane, int half, int ln,
                                                  const v16bf& aq0, const v16bf& aq1,
                                                  const v16bf kc[4], __bf16* pl,
                                                  v8f acc[4], float m[8], float l[8]) {
    const float scale = 0.125f;  // 1/sqrt(64)
    // V fragments: issued now, consumed after softmax (latency overlap)
    v16bf v0 = load_bT_frag(Vt, S, 0,  kk, lane);
    v16bf v1 = load_bT_frag(Vt, S, 16, kk, lane);
    v16bf v2 = load_bT_frag(Vt, S, 32, kk, lane);
    v16bf v3 = load_bT_frag(Vt, S, 48, kk, lane);

    // logits: two 16x16 tiles (keys kk..+15, kk+16..+31); HD=64 -> 2 K-steps
    v8f s0 = {}, s1 = {};
    s0 = WMMA_BF16(aq0, kc[0], s0);
    s0 = WMMA_BF16(aq1, kc[1], s0);
    s1 = WMMA_BF16(aq0, kc[2], s1);
    s1 = WMMA_BF16(aq1, kc[3], s1);

    // online softmax update; row r lives across the 16 lanes of this half
#pragma unroll
    for (int r = 0; r < 8; ++r) {
        float x0 = s0[r] * scale;
        float x1 = s1[r] * scale;
        float mx = fmaxf(x0, x1);
        mx = fmaxf(mx, __shfl_xor(mx, 1, 32));
        mx = fmaxf(mx, __shfl_xor(mx, 2, 32));
        mx = fmaxf(mx, __shfl_xor(mx, 4, 32));
        mx = fmaxf(mx, __shfl_xor(mx, 8, 32));
        const float mn = fmaxf(m[r], mx);
        const float alpha = __expf(m[r] - mn);
        const float p0 = __expf(x0 - mn);
        const float p1 = __expf(x1 - mn);
        float sum = p0 + p1;
        sum += __shfl_xor(sum, 1, 32);
        sum += __shfl_xor(sum, 2, 32);
        sum += __shfl_xor(sum, 4, 32);
        sum += __shfl_xor(sum, 8, 32);
        l[r] = l[r] * alpha + sum;
        m[r] = mn;
#pragma unroll
        for (int t = 0; t < 4; ++t) acc[t][r] *= alpha;
        // stage P (C-layout -> LDS row-major [16][32])
        const int prow = r + half * 8;
        pl[prow * 32 + ln]      = (__bf16)p0;
        pl[prow * 32 + 16 + ln] = (__bf16)p1;
    }

    // reload P in A-layout (DS ops are in-order within a wave)
    const v16bf pa = load_a_frag(pl, 32, 0, 0, lane);

    acc[0] = WMMA_BF16(pa, v0, acc[0]);
    acc[1] = WMMA_BF16(pa, v1, acc[1]);
    acc[2] = WMMA_BF16(pa, v2, acc[2]);
    acc[3] = WMMA_BF16(pa, v3, acc[3]);
}

static __device__ __forceinline__ void load_k4(v16bf kc[4], const __bf16* __restrict__ K,
                                               int kk, int lane) {
    kc[0] = load_bT_frag(K, HD, kk,      0,  lane);
    kc[1] = load_bT_frag(K, HD, kk,      32, lane);
    kc[2] = load_bT_frag(K, HD, kk + 16, 0,  lane);
    kc[3] = load_bT_frag(K, HD, kk + 16, 32, lane);
}

__global__ void __launch_bounds__(128, 1)
attn_kernel(const __bf16* __restrict__ qbuf,
            const __bf16* __restrict__ kbuf,
            const __bf16* __restrict__ vtbuf,
            __bf16* __restrict__ vals) {
    __shared__ __bf16 lds_p[4][16 * 32];     // one 16x32 P tile per wave

    const int lane  = threadIdx.x & 31;
    const int wslot = threadIdx.x >> 5;
    const int wave  = (blockIdx.x * blockDim.x + threadIdx.x) >> 5;  // 0..4095
    const int QT = S / 16;                   // 128 q-tiles per (b,h)
    const int bh = wave / QT;                // 0..31
    const int q0 = (wave % QT) * 16;

    const __bf16* Q  = qbuf  + (size_t)bh * S * HD;
    const __bf16* K  = kbuf  + (size_t)bh * S * HD;
    const __bf16* Vt = vtbuf + (size_t)bh * HD * S;   // [HD, S] row-major

    const v16bf aq0 = load_a_frag(Q, HD, q0, 0,  lane);
    const v16bf aq1 = load_a_frag(Q, HD, q0, 32, lane);

    v8f acc[4] = {};
    float m[8], l[8];
#pragma unroll
    for (int r = 0; r < 8; ++r) { m[r] = -1.0e30f; l[r] = 0.0f; }

    const int half = lane >> 4, ln = lane & 15;
    __bf16* pl = lds_p[wslot];

    // ping-pong over pairs of 32-key chunks
    v16bf kA[4], kB[4];
    load_k4(kA, K, 0, lane);
    for (int kk = 0; kk < S; kk += 64) {
        load_k4(kB, K, kk + 32, lane);                  // prefetch chunk kk+32
        attn_chunk(Vt, kk, lane, half, ln, aq0, aq1, kA, pl, acc, m, l);
        if (kk + 64 < S)
            load_k4(kA, K, kk + 64, lane);              // prefetch chunk kk+64
        attn_chunk(Vt, kk + 32, lane, half, ln, aq0, aq1, kB, pl, acc, m, l);
    }

    // epilogue: normalize, store bf16 values tile into [BS, D] layout
    const int b = bh / H, h = bh % H;
#pragma unroll
    for (int t = 0; t < 4; ++t) {
#pragma unroll
        for (int r = 0; r < 8; ++r) {
            const int row = q0 + r + half * 8;
            const float v = acc[t][r] / l[r];
            vals[((size_t)(b * S + row)) * D + h * HD + 16 * t + ln] = (__bf16)v;
        }
    }
}

// ---------------------------------------------------------------------------
// Kernel 4: output projection.  One wave per 32x64 tile of vals @ W_o^T + b_o.
// ---------------------------------------------------------------------------
__global__ void __launch_bounds__(128, 1)
out_gemm(const __bf16* __restrict__ vals,
         const __bf16* __restrict__ wo,
         const float* __restrict__ bias,
         float* __restrict__ out) {
    const int lane = threadIdx.x & 31;
    const int wave = (blockIdx.x * blockDim.x + threadIdx.x) >> 5;
    const int NT = D / 64;                   // 16
    const int mt = wave / NT;                // 0..127
    const int nt = wave % NT;                // 0..15
    const int m0 = mt * 32, n0 = nt * 64;

    v8f acc[2][4] = {};
    gemm_mainloop(vals, D, m0, wo, D, n0, D, lane, acc);

    const int half = lane >> 4, ln = lane & 15;
#pragma unroll
    for (int t = 0; t < 4; ++t) {
        const int n = n0 + 16 * t + ln;
        const float bv = bias[n];
#pragma unroll
        for (int msub = 0; msub < 2; ++msub) {
#pragma unroll
            for (int r = 0; r < 8; ++r) {
                const int row = m0 + msub * 16 + r + half * 8;
                out[(size_t)row * D + n] = acc[msub][t][r] + bv;
            }
        }
    }
}

// ---------------------------------------------------------------------------
// Host launcher
// ---------------------------------------------------------------------------
extern "C" void kernel_launch(void* const* d_in, const int* in_sizes, int n_in,
                              void* d_out, int out_size, void* d_ws, size_t ws_size,
                              hipStream_t stream) {
    const float* x    = (const float*)d_in[0];
    const float* Wqkv = (const float*)d_in[1];
    const float* bqkv = (const float*)d_in[2];
    const float* Wo   = (const float*)d_in[3];
    const float* bo   = (const float*)d_in[4];
    float* out = (float*)d_out;

    // Workspace layout (bf16 elements): 48 MiB total
    __bf16* ws    = (__bf16*)d_ws;
    __bf16* xb    = ws;                                   //  BS*D
    __bf16* wqkvb = xb    + (size_t)BS * D;               //  3D*D
    __bf16* wob   = wqkvb + (size_t)D3 * D;               //  D*D
    __bf16* qb    = wob   + (size_t)D * D;                //  BS*D
    __bf16* kb    = qb    + (size_t)BS * D;               //  BS*D
    __bf16* vtb   = kb    + (size_t)BS * D;               //  BS*D (transposed V)
    __bf16* valsb = vtb   + (size_t)BS * D;               //  BS*D

    const int nx = BS * D, nwq = D3 * D, nwo = D * D;
    cvt_f32_bf16<<<(nx / 8  + 255) / 256, 256, 0, stream>>>(x,    xb,    nx / 8);
    cvt_f32_bf16<<<(nwq / 8 + 255) / 256, 256, 0, stream>>>(Wqkv, wqkvb, nwq / 8);
    cvt_f32_bf16<<<(nwo / 8 + 255) / 256, 256, 0, stream>>>(Wo,   wob,   nwo / 8);

    // 128 m-tiles x 48 n-tiles = 6144 waves, 4 waves/block
    qkv_gemm<<<1536, 128, 0, stream>>>(xb, wqkvb, bqkv, qb, kb, vtb);

    // B*H*(S/16) = 4096 waves, 4 waves/block
    attn_kernel<<<1024, 128, 0, stream>>>(qb, kb, vtb, valsb);

    // 128 m-tiles x 16 n-tiles = 2048 waves
    out_gemm<<<512, 128, 0, stream>>>(valsb, wob, bo, out);
}
